// PointTransformerSeg_23287312679065
// MI455X (gfx1250) — compile-verified, matrix operand order checked by reading, not run on hardware
//
#include <hip/hip_runtime.h>
#include <hip/hip_bf16.h>
#include <math.h>

// ---------------------------------------------------------------------------
// Types for CDNA5 WMMA (wave32, 16x16x32 bf16 -> f32) and TDM descriptors
// ---------------------------------------------------------------------------
typedef __bf16 bf16;
typedef __attribute__((ext_vector_type(16))) __bf16 v16bf;
typedef __attribute__((ext_vector_type(8)))  float  v8f;
typedef __attribute__((ext_vector_type(4)))  unsigned v4u;
typedef __attribute__((ext_vector_type(8)))  int      v8i;
typedef __attribute__((ext_vector_type(4)))  int      v4i;

#if defined(__gfx1250__) && __has_builtin(__builtin_amdgcn_tensor_load_to_lds)
#define USE_TDM 1
#else
#define USE_TDM 0
#endif

#define GEMM_MT 64
#define GEMM_NT 64
#define GEMM_KT 32
#define GEMM_EPT 8            // (64*32)/256 elements staged per thread per tile
#define TILE_ELEMS (GEMM_NT * GEMM_KT)          // 2048 bf16 per weight tile
#define LDS_BYTES  ((GEMM_MT + GEMM_NT) * GEMM_KT * 2)

// ---------------------------------------------------------------------------
// Weight prepack: W (cout x cin, fp32) -> tiled bf16 [nt][kt][64][32], padded
// ---------------------------------------------------------------------------
__global__ void prepack_w_kernel(const float* __restrict__ W, int cout, int cin,
                                 bf16* __restrict__ Wp, int ntile, int ktile)
{
    int total = ntile * ktile * TILE_ELEMS;
    int t = blockIdx.x * blockDim.x + threadIdx.x;
    if (t >= total) return;
    int e = t & (TILE_ELEMS - 1);
    int tile = t >> 11;
    int ktn = tile % ktile;
    int ntn = tile / ktile;
    int row = e >> 5, kk = e & 31;
    int gr = ntn * GEMM_NT + row, gc = ktn * GEMM_KT + kk;
    float v = (gr < cout && gc < cin) ? W[(size_t)gr * cin + gc] : 0.0f;
    Wp[t] = (bf16)v;
}

// ---------------------------------------------------------------------------
// Fused GEMM: out = epilogue( A[n,cin] @ W[cout,cin]^T ), W prepacked bf16.
// A tile: fp32 -> bf16 staged by VALU.  B tile: 4KB contiguous blob pulled by
// the Tensor Data Mover (tensor_load_to_lds + s_wait_tensorcnt) when available.
// epilogue: (+bias[n]) -> (*g[n]*inv + B[n]) -> (+add[row,n]) -> relu
// ---------------------------------------------------------------------------
__global__ __launch_bounds__(256)
void wmma_gemm_kernel(const float* __restrict__ A, const bf16* __restrict__ Wp,
                      const float* __restrict__ bias, const float* __restrict__ bng,
                      const float* __restrict__ bnb, const float* __restrict__ addp,
                      float* __restrict__ out,
                      int nrows, int cin, int cout, int relu)
{
    extern __shared__ bf16 gemm_smem[];                 // dynamic LDS, offset 0
    bf16 (*As)[GEMM_KT] = (bf16(*)[GEMM_KT])gemm_smem;                 // 4KB
    bf16 (*Bs)[GEMM_KT] = (bf16(*)[GEMM_KT])(gemm_smem + GEMM_MT * GEMM_KT);
    const unsigned BS_LDS_OFF = GEMM_MT * GEMM_KT * sizeof(bf16);      // 4096

    const int tid  = threadIdx.x;
    const int lane = tid & 31;
    const int wave = tid >> 5;       // 8 waves
    const int half = lane >> 4;      // K-half selector (CDNA5 16-bit layout)
    const int l16  = lane & 15;
    const int wm   = wave & 3;       // 4 M sub-tiles of 16
    const int wn   = wave >> 2;      // 2 N groups of 32
    const int m0   = blockIdx.y * GEMM_MT;
    const int n0   = blockIdx.x * GEMM_NT;

    const int smm = tid >> 5;        // staging row phase base
    const int skk = tid & 31;        // staging k

    v8f acc0 = {};
    v8f acc1 = {};

    const int ksteps = (cin + GEMM_KT - 1) / GEMM_KT;
    for (int kt = 0; kt < ksteps; ++kt) {
        const int k0 = kt * GEMM_KT;
        const int gc = k0 + skk;
        const bool kin = (gc < cin);
        const size_t tilebase = ((size_t)blockIdx.x * ksteps + kt) * TILE_ELEMS;

        // ---- B tile: prepacked contiguous 4KB blob -> LDS ----
#if USE_TDM
        if (wave == 0) {
            // Tensor DMA descriptor (ISA 8.3/8.4): 1-row tile of 512 x 8B = 4KB
            unsigned long long ga =
                (unsigned long long)(size_t)(const void*)(Wp + tilebase);
            v4u g0 = {
                1u,                                               // count=1
                BS_LDS_OFF,                                       // lds_addr
                (unsigned)(ga & 0xffffffffu),                     // global_addr lo
                (unsigned)((ga >> 32) & 0x1ffffffu) | (2u << 30)  // addr hi | type=2
            };
            const unsigned tdim0 = 0x7fffffffu;   // no OOB clipping
            const unsigned tile0 = 512u;          // 512 * 8B = 4KB
            v8i g1 = {
                (int)0x00030000u,                   // data_size=3 (8B), flags=0
                (int)((tdim0 & 0xffffu) << 16),     // tensor_dim0[15:0]
                (int)((tdim0 >> 16) | (1u << 16)),  // tensor_dim0[31:16], tdim1=1
                (int)(tile0 << 16),                 // tile_dim0
                (int)1,                             // tile_dim1=1, tile_dim2=0
                (int)tile0,                         // tensor_dim0_stride
                0, 0
            };
            v4i g2 = {0, 0, 0, 0};
            v4i g3 = {0, 0, 0, 0};
            v8i g4 = {0, 0, 0, 0, 0, 0, 0, 0};
            __builtin_amdgcn_tensor_load_to_lds(g0, g1, g2, g3, g4, 0);
        }
#if __has_builtin(__builtin_amdgcn_s_wait_tensorcnt)
        __builtin_amdgcn_s_wait_tensorcnt(0);
#endif
#else
        {
            const bf16* wt = Wp + tilebase;
            bf16* bsf = &Bs[0][0];
            #pragma unroll
            for (int e = 0; e < GEMM_EPT; ++e) {
                int t = tid + e * 256;
                bsf[t] = wt[t];
            }
        }
#endif

        // ---- A tile: branch-free batched loads, fp32 -> bf16 ----
        float va[GEMM_EPT];
        bool  oa[GEMM_EPT];
        #pragma unroll
        for (int e = 0; e < GEMM_EPT; ++e) {
            int gra = m0 + smm + e * 8;
            oa[e] = (gra < nrows) & kin;
            size_t offa = oa[e] ? ((size_t)gra * cin + gc) : 0;
            va[e] = A[offa];
        }
        #pragma unroll
        for (int e = 0; e < GEMM_EPT; ++e) {
            As[smm + e * 8][skk] = (bf16)(oa[e] ? va[e] : 0.0f);
        }
        __syncthreads();

        // CDNA5 prefetch of next K-tile (emits global_prefetch_b8)
        if (kt + 1 < ksteps) {
            int pc = k0 + GEMM_KT;
            int gr = m0 + (tid & 63);
            if (gr < nrows && pc < cin) __builtin_prefetch(&A[(size_t)gr * cin + pc], 0, 1);
            if (tid == 0) __builtin_prefetch(Wp + tilebase + TILE_ELEMS, 0, 1);
        }

        // Assemble fragments per ISA 16-bit A/B layouts:
        //   lane<16: K-set {0..7,16..23}; lane>=16: K-set {8..15,24..31}
        union Frag { v16bf v; unsigned u[8]; };
        Frag fa, fb0, fb1;
        #pragma unroll
        for (int j = 0; j < 8; ++j) {
            int kb = ((j & 4) << 2) + (half << 3) + ((j & 3) << 1); // even -> aligned
            fa.u[j]  = *(const unsigned*)&As[wm * 16 + l16][kb];
            fb0.u[j] = *(const unsigned*)&Bs[wn * 32 + l16][kb];
            fb1.u[j] = *(const unsigned*)&Bs[wn * 32 + 16 + l16][kb];
        }
        acc0 = __builtin_amdgcn_wmma_f32_16x16x32_bf16(false, fa.v, false, fb0.v,
                                                       (short)0, acc0, false, false);
        acc1 = __builtin_amdgcn_wmma_f32_16x16x32_bf16(false, fa.v, false, fb1.v,
                                                       (short)0, acc1, false, false);
        __syncthreads();
    }

    // Epilogue on C/D fragment: VGPR r -> M = r + 8*half ; N = lane&15
    const float inv = rsqrtf(1.0f + 1e-5f);
    #pragma unroll
    for (int t = 0; t < 2; ++t) {
        v8f acc = t ? acc1 : acc0;
        int gn = n0 + wn * 32 + t * 16 + l16;
        #pragma unroll
        for (int r = 0; r < 8; ++r) {
            int gm = m0 + wm * 16 + r + half * 8;
            if (gm < nrows && gn < cout) {
                float y = acc[r];
                if (bias) y += bias[gn];
                if (bng)  y = y * (bng[gn] * inv) + bnb[gn];
                if (addp) y += addp[(size_t)gm * cout + gn];
                if (relu) y = fmaxf(y, 0.0f);
                out[(size_t)gm * cout + gn] = y;
            }
        }
    }
}

// ---------------------------------------------------------------------------
// Farthest point sampling: one persistent block, LDS argmax reduction
// ---------------------------------------------------------------------------
__global__ __launch_bounds__(1024)
void fps_kernel(const float* __restrict__ p, int n, int m,
                int* __restrict__ idx, float* __restrict__ dist)
{
    const int tid = threadIdx.x;
    const int nt  = blockDim.x;
    __shared__ float sval[1024];
    __shared__ int   sind[1024];
    __shared__ int   s_last;

    for (int i = tid; i < n; i += nt) dist[i] = 1e10f;
    if (tid == 0) { idx[0] = 0; s_last = 0; }
    __syncthreads();

    for (int it = 1; it < m; ++it) {
        int last = s_last;
        float lx = p[last * 3 + 0], ly = p[last * 3 + 1], lz = p[last * 3 + 2];
        float bv = -1.0f; int bi = 0x7fffffff;
        for (int i = tid; i < n; i += nt) {
            float dx = p[i * 3 + 0] - lx;
            float dy = p[i * 3 + 1] - ly;
            float dz = p[i * 3 + 2] - lz;
            float d  = dx * dx + dy * dy + dz * dz;
            float dd = fminf(dist[i], d);
            dist[i] = dd;
            if (dd > bv) { bv = dd; bi = i; }
        }
        sval[tid] = bv; sind[tid] = bi;
        __syncthreads();
        for (int s = nt >> 1; s > 0; s >>= 1) {
            if (tid < s) {
                if (sval[tid + s] > sval[tid] ||
                    (sval[tid + s] == sval[tid] && sind[tid + s] < sind[tid])) {
                    sval[tid] = sval[tid + s];
                    sind[tid] = sind[tid + s];
                }
            }
            __syncthreads();
        }
        if (tid == 0) { idx[it] = sind[0]; s_last = sind[0]; }
        __syncthreads();
    }
}

// ---------------------------------------------------------------------------
// kNN: per-query register top-K insertion; support tiled through LDS
// ---------------------------------------------------------------------------
template <int K>
__global__ __launch_bounds__(256)
void knn_kernel(const float* __restrict__ q, int nq,
                const float* __restrict__ s, int nsup,
                int* __restrict__ out)
{
    __shared__ float sp[256 * 3];
    const int i = blockIdx.x * blockDim.x + threadIdx.x;
    const bool active = (i < nq);
    float qx = 0.f, qy = 0.f, qz = 0.f;
    if (active) { qx = q[i * 3 + 0]; qy = q[i * 3 + 1]; qz = q[i * 3 + 2]; }

    float bd[K]; int bi[K];
    #pragma unroll
    for (int j = 0; j < K; ++j) { bd[j] = 3.4e38f; bi[j] = 0; }

    for (int base = 0; base < nsup; base += 256) {
        int tc = nsup - base; if (tc > 256) tc = 256;
        __syncthreads();
        if (threadIdx.x < tc) {
            sp[threadIdx.x * 3 + 0] = s[(size_t)(base + threadIdx.x) * 3 + 0];
            sp[threadIdx.x * 3 + 1] = s[(size_t)(base + threadIdx.x) * 3 + 1];
            sp[threadIdx.x * 3 + 2] = s[(size_t)(base + threadIdx.x) * 3 + 2];
        }
        __syncthreads();
        if (active) {
            for (int t = 0; t < tc; ++t) {
                float dx = sp[t * 3 + 0] - qx;
                float dy = sp[t * 3 + 1] - qy;
                float dz = sp[t * 3 + 2] - qz;
                float d  = dx * dx + dy * dy + dz * dz;
                if (d < bd[K - 1]) {
                    bd[K - 1] = d; bi[K - 1] = base + t;
                    #pragma unroll
                    for (int j = K - 1; j > 0; --j) {
                        if (bd[j] < bd[j - 1]) {
                            float tf = bd[j]; bd[j] = bd[j - 1]; bd[j - 1] = tf;
                            int   ti = bi[j]; bi[j] = bi[j - 1]; bi[j - 1] = ti;
                        }
                    }
                }
            }
        }
    }
    if (active) {
        #pragma unroll
        for (int j = 0; j < K; ++j) out[(size_t)i * K + j] = bi[j];
    }
}

// ---------------------------------------------------------------------------
// Small elementwise / gather kernels
// ---------------------------------------------------------------------------
__global__ void concat6_kernel(const float* __restrict__ p, const float* __restrict__ x,
                               int n, float* __restrict__ out)
{
    int t = blockIdx.x * blockDim.x + threadIdx.x;
    if (t >= n * 6) return;
    int i = t / 6, c = t % 6;
    out[t] = (c < 3) ? p[i * 3 + c] : x[i * 3 + (c - 3)];
}

__global__ void gather_rows3_kernel(const float* __restrict__ p, const int* __restrict__ idx,
                                    int m, float* __restrict__ out)
{
    int t = blockIdx.x * blockDim.x + threadIdx.x;
    if (t >= m * 3) return;
    int i = t / 3, c = t % 3;
    out[t] = p[(size_t)idx[i] * 3 + c];
}

__global__ void td_feats_kernel(const float* __restrict__ p, const float* __restrict__ x,
                                const float* __restrict__ pnew, const int* __restrict__ nn,
                                int mq, int ns, int cx, float* __restrict__ feats)
{
    int cf = 3 + cx;
    int t = blockIdx.x * blockDim.x + threadIdx.x;
    if (t >= mq * ns * cf) return;
    int ch = t % cf, rest = t / cf;
    int j = rest % ns, q = rest / ns;
    int pt = nn[(size_t)q * ns + j];
    feats[t] = (ch < 3) ? (p[(size_t)pt * 3 + ch] - pnew[(size_t)q * 3 + ch])
                        : x[(size_t)pt * cx + (ch - 3)];
}

__global__ void rowmax_kernel(const float* __restrict__ h, int mq, int ns, int c,
                              float* __restrict__ out)
{
    int t = blockIdx.x * blockDim.x + threadIdx.x;
    if (t >= mq * c) return;
    int q = t / c, ch = t % c;
    float m = -3.4e38f;
    for (int j = 0; j < ns; ++j) m = fmaxf(m, h[((size_t)q * ns + j) * c + ch]);
    out[t] = m;
}

__global__ void colmean_kernel(const float* __restrict__ x, int n, int c,
                               float* __restrict__ out)
{
    int ch = blockIdx.x * blockDim.x + threadIdx.x;
    if (ch >= c) return;
    float s = 0.f;
    for (int i = 0; i < n; ++i) s += x[(size_t)i * c + ch];
    out[ch] = s / (float)n;
}

__global__ void concat_g_kernel(const float* __restrict__ x5, const float* __restrict__ g,
                                int n, int c, float* __restrict__ out)
{
    int t = blockIdx.x * blockDim.x + threadIdx.x;
    if (t >= n * 2 * c) return;
    int i = t / (2 * c), ch = t % (2 * c);
    out[t] = (ch < c) ? x5[(size_t)i * c + ch] : g[ch - c];
}

__global__ void gather_pr_kernel(const float* __restrict__ p, const int* __restrict__ idx,
                                 int n, int ns, float* __restrict__ pr)
{
    int t = blockIdx.x * blockDim.x + threadIdx.x;
    if (t >= n * ns * 3) return;
    int ch = t % 3, rest = t / 3;
    int j = rest % ns, i = rest / ns;
    pr[t] = p[(size_t)idx[(size_t)i * ns + j] * 3 + ch] - p[(size_t)i * 3 + ch];
}

// w = relu( bn( xk[idx] - xq + pr ) )
__global__ void w_combine_kernel(const float* __restrict__ xk, const float* __restrict__ xq,
                                 const float* __restrict__ pr, const int* __restrict__ idx,
                                 const float* __restrict__ g, const float* __restrict__ B,
                                 int n, int ns, int c, float* __restrict__ w)
{
    int t = blockIdx.x * blockDim.x + threadIdx.x;
    if (t >= n * ns * c) return;
    int ch = t % c, rest = t / c;
    int j = rest % ns, i = rest / ns;
    const float inv = rsqrtf(1.0f + 1e-5f);
    float v = xk[(size_t)idx[(size_t)i * ns + j] * c + ch] - xq[(size_t)i * c + ch] + pr[t];
    v = v * (g[ch] * inv) + B[ch];
    w[t] = fmaxf(v, 0.0f);
}

__global__ void softmax_ns_kernel(float* __restrict__ w, int n, int ns, int cs)
{
    int t = blockIdx.x * blockDim.x + threadIdx.x;
    if (t >= n * cs) return;
    int i = t / cs, cp = t % cs;
    float mx = -3.4e38f;
    for (int j = 0; j < ns; ++j) mx = fmaxf(mx, w[((size_t)i * ns + j) * cs + cp]);
    float sum = 0.f;
    for (int j = 0; j < ns; ++j) {
        float e = expf(w[((size_t)i * ns + j) * cs + cp] - mx);
        w[((size_t)i * ns + j) * cs + cp] = e;
        sum += e;
    }
    float r = 1.0f / sum;
    for (int j = 0; j < ns; ++j) w[((size_t)i * ns + j) * cs + cp] *= r;
}

// out[i,ch] = sum_j (xv[idx[i,j],ch] + pr[i,j,ch]) * wsm[i,j,ch%cs]
__global__ void ptl_out_kernel(const float* __restrict__ xv, const float* __restrict__ pr,
                               const float* __restrict__ wsm, const int* __restrict__ idx,
                               int n, int ns, int c, float* __restrict__ out)
{
    int t = blockIdx.x * blockDim.x + threadIdx.x;
    if (t >= n * c) return;
    int i = t / c, ch = t % c;
    int cs = c >> 3;
    int cp = ch % cs;
    float acc = 0.f;
    for (int j = 0; j < ns; ++j) {
        int pt = idx[(size_t)i * ns + j];
        float v = xv[(size_t)pt * c + ch] + pr[((size_t)i * ns + j) * c + ch];
        acc += v * wsm[((size_t)i * ns + j) * cs + cp];
    }
    out[t] = acc;
}

__global__ void bn_relu_kernel(const float* __restrict__ x, const float* __restrict__ g,
                               const float* __restrict__ B, float* __restrict__ y,
                               int n, int c)
{
    int t = blockIdx.x * blockDim.x + threadIdx.x;
    if (t >= n * c) return;
    int ch = t % c;
    const float inv = rsqrtf(1.0f + 1e-5f);
    y[t] = fmaxf(x[t] * (g[ch] * inv) + B[ch], 0.0f);
}

// out = a + inverse-distance-weighted 3-NN interpolation of feat(src) at dst
__global__ void interp_add_kernel(const float* __restrict__ pdst, const float* __restrict__ psrc,
                                  const int* __restrict__ idx3, const float* __restrict__ feat,
                                  const float* __restrict__ a, int m, int c,
                                  float* __restrict__ out)
{
    int t = blockIdx.x * blockDim.x + threadIdx.x;
    if (t >= m * c) return;
    int i = t / c, ch = t % c;
    float px = pdst[i * 3 + 0], py = pdst[i * 3 + 1], pz = pdst[i * 3 + 2];
    float wv[3]; float wsum = 0.f;
    #pragma unroll
    for (int j = 0; j < 3; ++j) {
        int sidx = idx3[(size_t)i * 3 + j];
        float dx = px - psrc[sidx * 3 + 0];
        float dy = py - psrc[sidx * 3 + 1];
        float dz = pz - psrc[sidx * 3 + 2];
        float d = dx * dx + dy * dy + dz * dz;
        float w = 1.0f / (d + 1e-8f);
        wv[j] = w; wsum += w;
    }
    float acc = 0.f;
    #pragma unroll
    for (int j = 0; j < 3; ++j)
        acc += (wv[j] / wsum) * feat[(size_t)idx3[(size_t)i * 3 + j] * c + ch];
    out[t] = a[t] + acc;
}

// ---------------------------------------------------------------------------
// Host-side orchestration
// ---------------------------------------------------------------------------
static inline int cdiv(int a, int b) { return (a + b - 1) / b; }

struct Arena {
    char* base; size_t off;
    void* alloc(size_t bytes) {
        void* p = base + off;
        off = (off + bytes + 255) & ~(size_t)255;
        return p;
    }
    float* f(size_t n) { return (float*)alloc(n * sizeof(float)); }
    int*   i(size_t n) { return (int*)alloc(n * sizeof(int)); }
};

static const bf16* prep(hipStream_t s, Arena& pa, const float* W, int cout, int cin)
{
    int nt = cdiv(cout, GEMM_NT), kt = cdiv(cin, GEMM_KT);
    bf16* Wp = (bf16*)pa.alloc((size_t)nt * kt * TILE_ELEMS * sizeof(bf16));
    int total = nt * kt * TILE_ELEMS;
    prepack_w_kernel<<<cdiv(total, 256), 256, 0, s>>>(W, cout, cin, Wp, nt, kt);
    return Wp;
}

static void gemm(hipStream_t s, const float* A, const bf16* Wp, const float* bias,
                 const float* g, const float* B, const float* addp, float* out,
                 int n, int cin, int cout, bool relu)
{
    dim3 grid(cdiv(cout, GEMM_NT), cdiv(n, GEMM_MT));
    wmma_gemm_kernel<<<grid, 256, LDS_BYTES, s>>>(A, Wp, bias, g, B, addp, out,
                                                  n, cin, cout, relu ? 1 : 0);
}

struct Rd { void* const* d; int i; const float* nx() { return (const float*)d[i++]; } };

struct PtbP {
    const bf16 *l1_W, *q_W, *k_W, *v_W, *p1_W, *p2_W, *w1_W, *w2_W, *l3_W; // prepacked
    const float *bn1_g, *bn1_B;
    const float *q_b, *k_b, *v_b, *p1_b, *p2_b;
    const float *pbn_g, *pbn_B;
    const float *wbn1_g, *wbn1_B, *w1_b, *wbn2_g, *wbn2_B, *w2_b;
    const float *bn2_g, *bn2_B, *bn3_g, *bn3_B;
};

static PtbP read_ptb(Rd& r, hipStream_t s, Arena& pa, int c) {
    PtbP P; int cs = c / 8;
    P.l1_W = prep(s, pa, r.nx(), c, c);
    P.bn1_g = r.nx(); P.bn1_B = r.nx();
    P.q_W = prep(s, pa, r.nx(), c, c); P.q_b = r.nx();
    P.k_W = prep(s, pa, r.nx(), c, c); P.k_b = r.nx();
    P.v_W = prep(s, pa, r.nx(), c, c); P.v_b = r.nx();
    P.p1_W = prep(s, pa, r.nx(), 3, 3); P.p1_b = r.nx();
    P.pbn_g = r.nx(); P.pbn_B = r.nx();
    P.p2_W = prep(s, pa, r.nx(), c, 3); P.p2_b = r.nx();
    P.wbn1_g = r.nx(); P.wbn1_B = r.nx();
    P.w1_W = prep(s, pa, r.nx(), cs, c); P.w1_b = r.nx();
    P.wbn2_g = r.nx(); P.wbn2_B = r.nx();
    P.w2_W = prep(s, pa, r.nx(), cs, cs); P.w2_b = r.nx();
    P.bn2_g = r.nx(); P.bn2_B = r.nx();
    P.l3_W = prep(s, pa, r.nx(), c, c);
    P.bn3_g = r.nx(); P.bn3_B = r.nx();
    return P;
}

// Point-transformer block: out = relu( bn3(l3( ptl )) + x )
static void run_ptb(hipStream_t s, Arena sc, const PtbP& P, const float* p,
                    const float* x, int n, int c, int ns, float* out)
{
    float* h = sc.f((size_t)n * c);
    gemm(s, x, P.l1_W, nullptr, P.bn1_g, P.bn1_B, nullptr, h, n, c, c, true);

    float* xq = sc.f((size_t)n * c);
    float* xk = sc.f((size_t)n * c);
    float* xv = sc.f((size_t)n * c);
    gemm(s, h, P.q_W, P.q_b, nullptr, nullptr, nullptr, xq, n, c, c, false);
    gemm(s, h, P.k_W, P.k_b, nullptr, nullptr, nullptr, xk, n, c, c, false);
    gemm(s, h, P.v_W, P.v_b, nullptr, nullptr, nullptr, xv, n, c, c, false);

    int* idx = sc.i((size_t)n * ns);
    if (ns == 16) knn_kernel<16><<<cdiv(n, 256), 256, 0, s>>>(p, n, p, n, idx);
    else          knn_kernel<8> <<<cdiv(n, 256), 256, 0, s>>>(p, n, p, n, idx);

    float* prr = sc.f((size_t)n * ns * 3);
    gather_pr_kernel<<<cdiv(n * ns * 3, 256), 256, 0, s>>>(p, idx, n, ns, prr);
    float* pr1 = sc.f((size_t)n * ns * 3);
    gemm(s, prr, P.p1_W, P.p1_b, P.pbn_g, P.pbn_B, nullptr, pr1, n * ns, 3, 3, true);
    float* pr = sc.f((size_t)n * ns * c);
    gemm(s, pr1, P.p2_W, P.p2_b, nullptr, nullptr, nullptr, pr, n * ns, 3, c, false);

    float* w = sc.f((size_t)n * ns * c);
    w_combine_kernel<<<cdiv(n * ns * c, 256), 256, 0, s>>>(xk, xq, pr, idx,
                                                           P.wbn1_g, P.wbn1_B, n, ns, c, w);
    int cs = c / 8;
    float* w1 = sc.f((size_t)n * ns * cs);
    gemm(s, w, P.w1_W, P.w1_b, P.wbn2_g, P.wbn2_B, nullptr, w1, n * ns, c, cs, true);
    float* w2 = sc.f((size_t)n * ns * cs);
    gemm(s, w1, P.w2_W, P.w2_b, nullptr, nullptr, nullptr, w2, n * ns, cs, cs, false);
    softmax_ns_kernel<<<cdiv(n * cs, 256), 256, 0, s>>>(w2, n, ns, cs);

    float* po = sc.f((size_t)n * c);
    ptl_out_kernel<<<cdiv(n * c, 256), 256, 0, s>>>(xv, pr, w2, idx, n, ns, c, po);
    float* h2 = sc.f((size_t)n * c);
    bn_relu_kernel<<<cdiv(n * c, 256), 256, 0, s>>>(po, P.bn2_g, P.bn2_B, h2, n, c);

    // fused: bn3(l3(h2)) + x, relu  (residual + relu in WMMA epilogue)
    gemm(s, h2, P.l3_W, nullptr, P.bn3_g, P.bn3_B, x, out, n, c, c, true);
}

static void run_td(hipStream_t s, Arena sc, const float* p, const float* x,
                   int n, int cx, int cout,
                   const bf16* Wp, const float* g, const float* B,
                   float* pnew, float* xnew, int np1)
{
    float* dist = sc.f((size_t)n);
    int* fidx = sc.i((size_t)np1);
    fps_kernel<<<1, 1024, 0, s>>>(p, n, np1, fidx, dist);
    gather_rows3_kernel<<<cdiv(np1 * 3, 256), 256, 0, s>>>(p, fidx, np1, pnew);

    int* nn = sc.i((size_t)np1 * 16);
    knn_kernel<16><<<cdiv(np1, 256), 256, 0, s>>>(pnew, np1, p, n, nn);

    int cf = 3 + cx;
    float* feats = sc.f((size_t)np1 * 16 * cf);
    td_feats_kernel<<<cdiv(np1 * 16 * cf, 256), 256, 0, s>>>(p, x, pnew, nn, np1, 16, cx, feats);

    float* hh = sc.f((size_t)np1 * 16 * cout);
    gemm(s, feats, Wp, nullptr, g, B, nullptr, hh, np1 * 16, cf, cout, true);
    rowmax_kernel<<<cdiv(np1 * cout, 256), 256, 0, s>>>(hh, np1, 16, cout, xnew);
}

extern "C" void kernel_launch(void* const* d_in, const int* in_sizes, int n_in,
                              void* d_out, int out_size, void* d_ws, size_t ws_size,
                              hipStream_t stream)
{
    (void)in_sizes; (void)n_in; (void)out_size; (void)ws_size;
    const int N = 16384;
    const float* p0 = (const float*)d_in[0];
    const float* x0 = (const float*)d_in[1];
    // d_in[2] = o0 (unused)

    // Workspace: persistent region (level tensors + prepacked weights) + scratch
    Arena persist{(char*)d_ws, 0};
    Arena scratch0{(char*)d_ws + ((size_t)64 << 20), 0};

    const int n1 = N, n2 = 4096, n3 = 1024, n4 = 256, n5 = 64;
    const int c1 = 32, c2 = 64, c3 = 128, c4 = 256, c5 = 512;
    const int encCi[4] = {35, 67, 131, 259};
    const int encCo[4] = {c2, c3, c4, c5};

    Rd r{d_in, 3};
    const bf16* enc1_W = prep(stream, persist, r.nx(), c1, 6);
    const float* enc1_g = r.nx(); const float* enc1_B = r.nx();
    const bf16 *encW[4]; const float *encg[4], *encB[4];
    for (int i = 0; i < 4; ++i) {
        encW[i] = prep(stream, persist, r.nx(), encCo[i], encCi[i]);
        encg[i] = r.nx(); encB[i] = r.nx();
    }
    const bf16* d5l1_W = prep(stream, persist, r.nx(), c5, 2 * c5);
    const float* d5l1_b = r.nx();
    const float* d5bn1_g = r.nx(); const float* d5bn1_B = r.nx();
    const bf16* d5l2_W = prep(stream, persist, r.nx(), c5, c5);
    const float* d5l2_b = r.nx();
    PtbP d5b = read_ptb(r, stream, persist, c5);
    struct DecP { const bf16 *l1_W, *l2_W; const float *l1_b, *bn1_g, *bn1_B,
                  *l2_b, *bn2_g, *bn2_B; PtbP b; } dec[4];
    const int decC[4]   = {c4, c3, c2, c1};
    const int decCin[4] = {c5, c4, c3, c2};
    for (int i = 0; i < 4; ++i) { // dec4, dec3, dec2, dec1
        dec[i].l1_W = prep(stream, persist, r.nx(), decC[i], decC[i]);
        dec[i].l1_b = r.nx();
        dec[i].bn1_g = r.nx(); dec[i].bn1_B = r.nx();
        dec[i].l2_W = prep(stream, persist, r.nx(), decC[i], decCin[i]);
        dec[i].l2_b = r.nx();
        dec[i].bn2_g = r.nx(); dec[i].bn2_B = r.nx();
        dec[i].b = read_ptb(r, stream, persist, decC[i]);
    }
    const bf16* cls1_W = prep(stream, persist, r.nx(), 32, 32);
    const float* cls1_b = r.nx();
    const float* clsbn_g = r.nx(); const float* clsbn_B = r.nx();
    const bf16* cls2_W = prep(stream, persist, r.nx(), 13, 32);
    const float* cls2_b = r.nx();

    float* x  = persist.f((size_t)N * 6);
    float* x1 = persist.f((size_t)n1 * c1);
    float* p2 = persist.f((size_t)n2 * 3);  float* x2 = persist.f((size_t)n2 * c2);
    float* p3 = persist.f((size_t)n3 * 3);  float* x3 = persist.f((size_t)n3 * c3);
    float* p4 = persist.f((size_t)n4 * 3);  float* x4 = persist.f((size_t)n4 * c4);
    float* p5 = persist.f((size_t)n5 * 3);  float* x5 = persist.f((size_t)n5 * c5);
    float* h5 = persist.f((size_t)n5 * c5);
    float* h4 = persist.f((size_t)n4 * c4);
    float* h3 = persist.f((size_t)n3 * c3);
    float* h2 = persist.f((size_t)n2 * c2);
    float* h1 = persist.f((size_t)n1 * c1);

    // ---- encoder ----
    concat6_kernel<<<cdiv(N * 6, 256), 256, 0, stream>>>(p0, x0, N, x);
    gemm(stream, x, enc1_W, nullptr, enc1_g, enc1_B, nullptr, x1, N, 6, c1, true);

    run_td(stream, scratch0, p0, x1, n1, c1, c2, encW[0], encg[0], encB[0], p2, x2, n2);
    run_td(stream, scratch0, p2, x2, n2, c2, c3, encW[1], encg[1], encB[1], p3, x3, n3);
    run_td(stream, scratch0, p3, x3, n3, c3, c4, encW[2], encg[2], encB[2], p4, x4, n4);
    run_td(stream, scratch0, p4, x4, n4, c4, c5, encW[3], encg[3], encB[3], p5, x5, n5);

    // ---- dec5 (bottleneck) ----
    {
        Arena sc = scratch0;
        float* mu = sc.f(c5);
        colmean_kernel<<<cdiv(c5, 256), 256, 0, stream>>>(x5, n5, c5, mu);
        float* g = sc.f(c5);
        gemm(stream, mu, d5l2_W, d5l2_b, nullptr, nullptr, nullptr, g, 1, c5, c5, true);
        float* h5cat = sc.f((size_t)n5 * 2 * c5);
        concat_g_kernel<<<cdiv(n5 * 2 * c5, 256), 256, 0, stream>>>(x5, g, n5, c5, h5cat);
        float* h5in = sc.f((size_t)n5 * c5);
        gemm(stream, h5cat, d5l1_W, d5l1_b, d5bn1_g, d5bn1_B, nullptr, h5in, n5, 2 * c5, c5, true);
        run_ptb(stream, sc, d5b, p5, h5in, n5, c5, 16, h5);
    }

    // ---- decoder stages ----
    struct Lvl { const float *p_hi, *x_hi; int n_hi; const float *p_lo, *x_lo; int n_lo;
                 int c; int cin_lo; int ns; float* out; };
    Lvl lv[4] = {
        { p4, x4, n4, p5, h5, n5, c4, c5, 16, h4 },
        { p3, x3, n3, p4, h4, n4, c3, c4, 16, h3 },
        { p2, x2, n2, p3, h3, n3, c2, c3, 16, h2 },
        { p0, x1, n1, p2, h2, n2, c1, c2,  8, h1 },
    };
    for (int i = 0; i < 4; ++i) {
        Arena sc = scratch0;
        const Lvl& L = lv[i];
        float* a = sc.f((size_t)L.n_hi * L.c);
        gemm(stream, L.x_hi, dec[i].l1_W, dec[i].l1_b, dec[i].bn1_g, dec[i].bn1_B,
             nullptr, a, L.n_hi, L.c, L.c, true);
        float* b = sc.f((size_t)L.n_lo * L.c);
        gemm(stream, L.x_lo, dec[i].l2_W, dec[i].l2_b, dec[i].bn2_g, dec[i].bn2_B,
             nullptr, b, L.n_lo, L.cin_lo, L.c, true);
        int* idx3 = sc.i((size_t)L.n_hi * 3);
        knn_kernel<3><<<cdiv(L.n_hi, 256), 256, 0, stream>>>(L.p_hi, L.n_hi, L.p_lo, L.n_lo, idx3);
        float* xin = sc.f((size_t)L.n_hi * L.c);
        interp_add_kernel<<<cdiv(L.n_hi * L.c, 256), 256, 0, stream>>>(
            L.p_hi, L.p_lo, idx3, b, a, L.n_hi, L.c, xin);
        run_ptb(stream, sc, dec[i].b, L.p_hi, xin, L.n_hi, L.c, L.ns, L.out);
    }

    // ---- classifier head ----
    {
        Arena sc = scratch0;
        float* tmp = sc.f((size_t)N * 32);
        gemm(stream, h1, cls1_W, cls1_b, clsbn_g, clsbn_B, nullptr, tmp, N, 32, 32, true);
        gemm(stream, tmp, cls2_W, cls2_b, nullptr, nullptr, nullptr,
             (float*)d_out, N, 32, 13, false);
    }
}
